// SAGENet_47330539602644
// MI455X (gfx1250) — matrix operand order repaired and checked by reading, not compiled, not run on hardware
//
#include <hip/hip_runtime.h>
#include <hip/hip_bf16.h>

typedef __attribute__((ext_vector_type(16))) _Float16 v16h;
typedef __attribute__((ext_vector_type(8)))  _Float16 v8h;
typedef __attribute__((ext_vector_type(8)))  float    v8f;

#define NNODES 50000

// ---------------- utility kernels ----------------

__global__ void fill_zero(float* __restrict__ p, size_t n) {
  size_t i = (size_t)blockIdx.x * blockDim.x + threadIdx.x;
  if (i < n) p[i] = 0.0f;
}

__global__ void cvt_f32_to_f16(const float* __restrict__ in, _Float16* __restrict__ out, size_t n) {
  size_t i = (size_t)blockIdx.x * blockDim.x + threadIdx.x;
  if (i < n) out[i] = (_Float16)in[i];
}

__global__ void degree_kernel(const int* __restrict__ dst, float* __restrict__ deg, int E) {
  int e = blockIdx.x * blockDim.x + threadIdx.x;
  if (e < E) atomicAdd(&deg[dst[e]], 1.0f);
}

// gather x[src] -> atomic-add into agg[dst]; (F/4) threads per edge, float4 gathers
__global__ void scatter_add_kernel(const float* __restrict__ x,
                                   const int* __restrict__ src,
                                   const int* __restrict__ dst,
                                   float* __restrict__ agg,
                                   int E, int F, int logTpe) {
  long long tid = (long long)blockIdx.x * blockDim.x + threadIdx.x;
  long long e = tid >> logTpe;
  if (e >= E) return;
  int r = (int)(tid & ((1 << logTpe) - 1));
  int s = src[e], d = dst[e];
  const float4 v = *reinterpret_cast<const float4*>(x + (size_t)s * F + r * 4);
  float* o = agg + (size_t)d * F + r * 4;
  atomicAdd(o + 0, v.x);
  atomicAdd(o + 1, v.y);
  atomicAdd(o + 2, v.z);
  atomicAdd(o + 3, v.w);
}

// out[i] = f16( agg[i] / max(deg[i>>logF],1) )
__global__ void mean_div_cvt_kernel(const float* __restrict__ agg, const float* __restrict__ deg,
                                    _Float16* __restrict__ out, int logF, size_t total) {
  size_t i = (size_t)blockIdx.x * blockDim.x + threadIdx.x;
  if (i < total) out[i] = (_Float16)(agg[i] / fmaxf(deg[i >> logF], 1.0f));
}

// ---------------- WMMA GEMM (pure f16 operands, f32 accumulate) ----------------

// A fragment (16x32 f16): lane m=lane&15, hi=lane>>4; 16 consecutive K at k+hi*16
__device__ __forceinline__ v16h load_a_frag16(const _Float16* __restrict__ p) {
  v8h lo = *reinterpret_cast<const v8h*>(p);       // k+hi*16 .. +7
  v8h hiv = *reinterpret_cast<const v8h*>(p + 8);  // k+hi*16+8 .. +7
  v16h r;
#pragma unroll
  for (int i = 0; i < 8; ++i) { r[i] = lo[i]; r[8 + i] = hiv[i]; }
  return r;
}

// B fragment (32x16 f16): lane n=lane&15, hi=lane>>4; 16 consecutive K at k+hi*16
__device__ __forceinline__ v16h load_b_frag16(const _Float16* __restrict__ p) {
  v8h lo = *reinterpret_cast<const v8h*>(p);
  v8h hiv = *reinterpret_cast<const v8h*>(p + 8);
  v16h r;
#pragma unroll
  for (int i = 0; i < 8; ++i) { r[i] = lo[i]; r[8 + i] = hiv[i]; }
  return r;
}

// outF[M,N] = relu?( A1@B1^T + A2@B2^T + bias ); optional f16 shadow copy in outH.
// One wave computes a 16x64 tile (4 N-tiles). B1,B2 are [N,K] row-major. N%64==0, M%16==0, K%32==0.
__global__ void sage_gemm_f16(const _Float16* __restrict__ A1, const _Float16* __restrict__ A2,
                              const _Float16* __restrict__ B1, const _Float16* __restrict__ B2,
                              const float* __restrict__ bias,
                              float* __restrict__ outF, _Float16* __restrict__ outH,
                              int M, int N, int K, int doRelu) {
  int wave = blockIdx.x * (blockDim.x >> 5) + (threadIdx.x >> 5);
  int lane = threadIdx.x & 31;
  int Ngr = N >> 6;               // groups of 64 output columns
  int mt = wave / Ngr;            // wave-uniform -> EXEC all-ones at WMMA
  int ng = wave - mt * Ngr;
  if (mt >= (M >> 4)) return;

  int m0 = mt << 4, n0 = ng << 6;
  int ml = lane & 15;
  int hi = lane >> 4;

  const _Float16* a1 = A1 + (size_t)(m0 + ml) * K + hi * 16;
  const _Float16* a2 = A2 + (size_t)(m0 + ml) * K + hi * 16;
  const _Float16* b1 = B1 + (size_t)(n0 + ml) * K + hi * 16;
  const _Float16* b2 = B2 + (size_t)(n0 + ml) * K + hi * 16;
  const size_t bstride = (size_t)16 * K;  // next N-tile

  v8f acc[4] = {};
  for (int k = 0; k < K; k += 32) {
    v16h fa1 = load_a_frag16(a1 + k);
    v16h fa2 = load_a_frag16(a2 + k);
#pragma unroll
    for (int t = 0; t < 4; ++t) {
      v16h fb1 = load_b_frag16(b1 + t * bstride + k);
      v16h fb2 = load_b_frag16(b2 + t * bstride + k);
      acc[t] = __builtin_amdgcn_wmma_f32_16x16x32_f16(false, fa1, false, fb1,
                                                      (short)0, acc[t], false, false);
      acc[t] = __builtin_amdgcn_wmma_f32_16x16x32_f16(false, fa2, false, fb2,
                                                      (short)0, acc[t], false, false);
    }
  }

#pragma unroll
  for (int t = 0; t < 4; ++t) {
    int nn = n0 + (t << 4) + ml;
    float b = bias[nn];
#pragma unroll
    for (int v = 0; v < 8; ++v) {
      int m = m0 + v + (hi << 3);      // C/D layout: VGPR v -> row v (+8 for hi lanes)
      float r = acc[t][v] + b;
      if (doRelu) r = fmaxf(r, 0.0f);
      outF[(size_t)m * N + nn] = r;
      if (outH) outH[(size_t)m * N + nn] = (_Float16)r;
    }
  }
}

// ---------------- launch ----------------

extern "C" void kernel_launch(void* const* d_in, const int* in_sizes, int n_in,
                              void* d_out, int out_size, void* d_ws, size_t ws_size,
                              hipStream_t stream) {
  const float* x   = (const float*)d_in[0];
  const int*   ei  = (const int*)d_in[1];   // [2,E] int32
  const float* W1l = (const float*)d_in[2];
  const float* b1l = (const float*)d_in[3];
  const float* W1r = (const float*)d_in[4];
  const float* W2l = (const float*)d_in[5];
  const float* b2l = (const float*)d_in[6];
  const float* W2r = (const float*)d_in[7];
  float* out = (float*)d_out;

  const int N = NNODES, Fin = 128, H1 = 256, H2 = 128;
  const int E = in_sizes[1] / 2;
  const int* src = ei;
  const int* dst = ei + E;

  // f32 workspace: deg[pad 50048] | agg1[N*Fin] | h[N*H1] | agg2[N*H1]
  float* ws   = (float*)d_ws;
  float* deg  = ws;
  float* agg1 = ws + 50048;
  float* h    = agg1 + (size_t)N * Fin;
  float* agg2 = h    + (size_t)N * H1;
  // f16 workspace after the f32 region: xh | m1h | hh | m2h | w1l | w1r | w2l | w2r
  _Float16* fp16base = (_Float16*)(agg2 + (size_t)N * H1);
  _Float16* xh  = fp16base;
  _Float16* m1h = xh  + (size_t)N * Fin;
  _Float16* hh  = m1h + (size_t)N * Fin;
  _Float16* m2h = hh  + (size_t)N * H1;
  _Float16* w1lh = m2h  + (size_t)N * H1;
  _Float16* w1rh = w1lh + (size_t)H1 * Fin;
  _Float16* w2lh = w1rh + (size_t)H1 * Fin;
  _Float16* w2rh = w2lh + (size_t)H2 * H1;

  const int BT = 256;

  // zero deg + agg1 (contiguous)
  size_t z1 = 50048 + (size_t)N * Fin;
  fill_zero<<<(unsigned)((z1 + BT - 1) / BT), BT, 0, stream>>>(ws, z1);

  // degrees (same dst for both layers)
  degree_kernel<<<(E + BT - 1) / BT, BT, 0, stream>>>(dst, deg, E);

  // one-time f16 conversions: x and all four weight matrices
  size_t nx = (size_t)N * Fin;
  cvt_f32_to_f16<<<(unsigned)((nx + BT - 1) / BT), BT, 0, stream>>>(x, xh, nx);
  size_t nw1 = (size_t)H1 * Fin, nw2 = (size_t)H2 * H1;
  cvt_f32_to_f16<<<(unsigned)((nw1 + BT - 1) / BT), BT, 0, stream>>>(W1l, w1lh, nw1);
  cvt_f32_to_f16<<<(unsigned)((nw1 + BT - 1) / BT), BT, 0, stream>>>(W1r, w1rh, nw1);
  cvt_f32_to_f16<<<(unsigned)((nw2 + BT - 1) / BT), BT, 0, stream>>>(W2l, w2lh, nw2);
  cvt_f32_to_f16<<<(unsigned)((nw2 + BT - 1) / BT), BT, 0, stream>>>(W2r, w2rh, nw2);

  // layer 1 aggregate: tpe = Fin/4 = 32 -> logTpe 5
  {
    long long T = (long long)E * (Fin >> 2);
    scatter_add_kernel<<<(unsigned)((T + BT - 1) / BT), BT, 0, stream>>>(x, src, dst, agg1, E, Fin, 5);
  }
  mean_div_cvt_kernel<<<(unsigned)((nx + BT - 1) / BT), BT, 0, stream>>>(agg1, deg, m1h, 7, nx);

  // layer 1 GEMM: h = relu(mean1@W1l^T + b1l + x@W1r^T), also emit f16 shadow hh
  {
    int waves = (N >> 4) * (H1 >> 6);   // 3125 * 4
    sage_gemm_f16<<<(waves + 7) / 8, BT, 0, stream>>>(m1h, xh, w1lh, w1rh, b1l, h, hh, N, H1, Fin, 1);
  }

  // layer 2 aggregate
  size_t z2 = (size_t)N * H1;
  fill_zero<<<(unsigned)((z2 + BT - 1) / BT), BT, 0, stream>>>(agg2, z2);
  {
    long long T = (long long)E * (H1 >> 2);  // tpe = 64 -> logTpe 6
    scatter_add_kernel<<<(unsigned)((T + BT - 1) / BT), BT, 0, stream>>>(h, src, dst, agg2, E, H1, 6);
  }
  mean_div_cvt_kernel<<<(unsigned)((z2 + BT - 1) / BT), BT, 0, stream>>>(agg2, deg, m2h, 8, z2);

  // layer 2 GEMM: out = mean2@W2l^T + b2l + h@W2r^T
  {
    int waves = (N >> 4) * (H2 >> 6);   // 3125 * 2
    sage_gemm_f16<<<(waves + 7) / 8, BT, 0, stream>>>(m2h, hh, w2lh, w2rh, b2l, out, (_Float16*)nullptr, N, H2, H1, 0);
  }
}
